// KalmanCNN_73847667687641
// MI455X (gfx1250) — matrix operand order repaired and checked by reading, not compiled, not run on hardware
//
#include <hip/hip_runtime.h>
#include <hip/hip_bf16.h>
#include <math.h>

typedef __attribute__((ext_vector_type(16))) _Float16 v16h;
typedef __attribute__((ext_vector_type(8)))  _Float16 v8h;
typedef __attribute__((ext_vector_type(2)))  _Float16 h2v;
typedef __attribute__((ext_vector_type(8)))  float    v8f;

struct FuseParams {
  const float* inputy;   // (B,2,11)
  const float* target;   // (B,4,11)
  const float* Hm;       // (2,4)
  const float* gains;    // 10 x {dt, v, K[4][2]}
  const _Float16* wf[14];
  const float*    bs[14];
  float* out;            // (B,8)
};

// -------------------------------------------------------------------------
// Kalman gain precompute: P/S/K recursion is batch-independent (F,Q,H,R fixed
// per step, P0 = m2x). One thread, 10 steps of 4x4 math.
// -------------------------------------------------------------------------
__global__ void kalman_gains_kernel(const float* __restrict__ t_fa,
                                    const float* __restrict__ v_fa,
                                    const float* __restrict__ m2x,
                                    const float* __restrict__ Hm,
                                    const float* __restrict__ Qm,
                                    const float* __restrict__ Rm,
                                    float* __restrict__ gout)
{
  if (blockIdx.x != 0 || threadIdx.x != 0) return;
  float P[4][4], H[2][4];
  for (int r = 0; r < 4; ++r) for (int c = 0; c < 4; ++c) P[r][c] = m2x[r*4+c];
  for (int r = 0; r < 2; ++r) for (int c = 0; c < 4; ++c) H[r][c] = Hm[r*4+c];
  for (int i = 0; i < 10; ++i) {
    float dt = t_fa[1002+i] - t_fa[1001+i];
    float v  = v_fa[1001+i];
    float F[4][4] = {{1.f,-dt,-v*dt,-v*v*dt},{0.f,1.f,0.f,0.f},{0.f,0.f,1.f,0.f},{0.f,0.f,0.f,1.f}};
    float FP[4][4], Pp[4][4];
    for (int r = 0; r < 4; ++r) for (int c = 0; c < 4; ++c) {
      float s = 0.f; for (int k = 0; k < 4; ++k) s += F[r][k]*P[k][c]; FP[r][c] = s;
    }
    for (int r = 0; r < 4; ++r) for (int c = 0; c < 4; ++c) {
      float s = 0.f; for (int k = 0; k < 4; ++k) s += FP[r][k]*F[c][k]; Pp[r][c] = s + Qm[r*4+c];
    }
    float HP[2][4], S[2][2], T[4][2];
    for (int r = 0; r < 2; ++r) for (int c = 0; c < 4; ++c) {
      float s = 0.f; for (int k = 0; k < 4; ++k) s += H[r][k]*Pp[k][c]; HP[r][c] = s;
    }
    for (int r = 0; r < 2; ++r) for (int c = 0; c < 2; ++c) {
      float s = 0.f; for (int k = 0; k < 4; ++k) s += HP[r][k]*H[c][k]; S[r][c] = s + Rm[r*2+c];
    }
    for (int r = 0; r < 4; ++r) for (int c = 0; c < 2; ++c) {
      float s = 0.f; for (int k = 0; k < 4; ++k) s += Pp[r][k]*H[c][k]; T[r][c] = s;
    }
    float det = S[0][0]*S[1][1] - S[0][1]*S[1][0];
    float id  = 1.f / det;
    float Si[2][2] = {{ S[1][1]*id, -S[0][1]*id}, {-S[1][0]*id,  S[0][0]*id}};
    float K[4][2], KS[4][2];
    for (int r = 0; r < 4; ++r) for (int c = 0; c < 2; ++c)
      K[r][c] = T[r][0]*Si[0][c] + T[r][1]*Si[1][c];
    for (int r = 0; r < 4; ++r) for (int c = 0; c < 2; ++c)
      KS[r][c] = K[r][0]*S[0][c] + K[r][1]*S[1][c];
    for (int r = 0; r < 4; ++r) for (int c = 0; c < 4; ++c)
      P[r][c] = Pp[r][c] - (KS[r][0]*K[c][0] + KS[r][1]*K[c][1]);
    gout[i*10+0] = dt; gout[i*10+1] = v;
    for (int m = 0; m < 4; ++m) for (int j = 0; j < 2; ++j) gout[i*10+2+m*2+j] = K[m][j];
  }
}

// -------------------------------------------------------------------------
// Weight swizzle: fp32 OIH / (out,in) weights -> f16 WMMA A-fragments.
// A-layout used here (contiguous k, not the ISA table order; 16-bit A 16x32): row M = lane&15,
//   K(lane,h) = h + (lane>=16 ? 16 : 0).
// GEMM-K ordering: q = k*Cin + ic. Zero-padded outside [0, Cin*ksz).
// Fragment storage: dst[((ot*ktiles + kt)*32 + lane)*16 + h] -> one v16h load.
// -------------------------------------------------------------------------
__global__ void swz_kernel(const float* __restrict__ w, _Float16* __restrict__ dst,
                           int Cin, int Cout, int ksz, int ktiles, int total)
{
  int tid = blockIdx.x*blockDim.x + threadIdx.x;
  if (tid >= total) return;
  int h    = tid & 15;
  int lane = (tid >> 4) & 31;
  int fr   = tid >> 9;
  int kt   = fr % ktiles;
  int ot   = fr / ktiles;
  int oc   = ot*16 + (lane & 15);
  int q    = kt*32 + h + ((lane >= 16) ? 16 : 0);
  float val = 0.f;
  if (oc < Cout && q < Cin*ksz) {
    int k  = (q >= Cin) ? 1 : 0;   // ksz <= 2
    int ic = q - k*Cin;
    val = w[(oc*Cin + ic)*ksz + k];
  }
  dst[tid] = (_Float16)val;
}

// -------------------------------------------------------------------------
// Implicit-GEMM conv on v_wmma_f32_16x16x32_f16.
// LDS activation layout: act[n][t*C + ic]  (row stride C*L halves per column).
// With q = k*Cin+ic, the K-slice for output position p is act[n][p*C + q]:
// CONTIGUOUS per lane -> single aligned v16h LDS load when Cin%16==0.
// Small layers drop the q<Ktot guard: over-read hits zero-initialized /
// finite LDS and multiplies zero-padded weights (contributes exactly 0).
// D store: packed v8h (consecutive oc) when layout allows, else scalar.
// -------------------------------------------------------------------------
template<int Cin, int Cout, int Lin, int Ksz, bool Relu, int OutStride = -1>
__device__ inline void conv_wmma(const _Float16* __restrict__ wf,
                                 const float* __restrict__ bias,
                                 const _Float16* in, _Float16* out, int lane)
{
  constexpr int Ktot = Cin*Ksz;
  constexpr int KT   = (Ktot + 31) / 32;
  constexpr int OT   = (Cout + 15) / 16;
  constexpr int Lc   = Lin - (Ksz - 1);
  constexpr int OS   = (OutStride < 0) ? Cout*Lc : OutStride;
  constexpr bool VLD = (Cin % 16) == 0;                    // 32B-aligned B load
  constexpr bool VST = ((Cout % 16) == 0) && ((OS % 8) == 0); // 16B-aligned D store
  const int n     = lane & 15;
  const int rbase = (lane & 16) ? 8 : 0;
  const int qoff  = (lane & 16) ? 16 : 0;
  const int inrow = n*(Cin*Lin);
  for (int p = 0; p < Lc; ++p) {
    for (int ot = 0; ot < OT; ++ot) {
      v8f acc = {0.f,0.f,0.f,0.f,0.f,0.f,0.f,0.f};
      for (int kt = 0; kt < KT; ++kt) {
        v16h a = *(const v16h*)(wf + (size_t)((ot*KT + kt)*32 + lane)*16);
        v16h bfrag;
        const int base = inrow + p*Cin + kt*32 + qoff;
        if constexpr (VLD) {
          bfrag = *(const v16h*)(in + base);
        } else {
#pragma unroll
          for (int h = 0; h < 16; ++h) bfrag[h] = in[base + h];
        }
        acc = __builtin_amdgcn_wmma_f32_16x16x32_f16(false, a, false, bfrag,
                                                     (short)0, acc, false, false);
      }
      if constexpr (VST) {
        v8h o;
#pragma unroll
        for (int r = 0; r < 8; ++r) {
          float v = acc[r] + bias[ot*16 + rbase + r];
          if (Relu) v = (v > 0.f) ? v : 0.f;
          o[r] = (_Float16)v;
        }
        *(v8h*)(out + n*OS + p*Cout + ot*16 + rbase) = o;
      } else {
#pragma unroll
        for (int r = 0; r < 8; ++r) {
          int oc = ot*16 + rbase + r;
          if (oc < Cout) {
            float v = acc[r] + bias[oc];
            if (Relu) v = (v > 0.f) ? v : 0.f;
            out[n*OS + p*Cout + oc] = (_Float16)v;
          }
        }
      }
    }
  }
}

// maxpool window 2, stride 1 (VALID) over the length axis; packed half-pairs.
template<int C, int Lc>
__device__ inline void maxpool_lds(const _Float16* in, _Float16* out, int lane)
{
  constexpr int rowin  = C*Lc;
  constexpr int rowout = C*(Lc - 1);
  constexpr int rp     = rowout/2;     // pairs per column (C even everywhere)
  constexpr int total2 = 16*rp;
  for (int idx = lane; idx < total2; idx += 32) {
    int nn = idx / rp;
    int j  = idx - nn*rp;
    int e  = 2*j;
    int p  = e / C;
    int c  = e - p*C;
    h2v a = *(const h2v*)(in + nn*rowin + p*C + c);
    h2v b = *(const h2v*)(in + nn*rowin + (p+1)*C + c);
    h2v o;
    o[0] = ((float)a[0] >= (float)b[0]) ? a[0] : b[0];
    o[1] = ((float)a[1] >= (float)b[1]) ? a[1] : b[1];
    *(h2v*)(out + nn*rowout + e) = o;
  }
}

__device__ inline void normv(float* v, int nc)
{
  float s = 0.f;
  for (int i = 0; i < nc; ++i) s += v[i]*v[i];
  float nr = fmaxf(sqrtf(s), 1e-12f);
  for (int i = 0; i < nc; ++i) v[i] /= nr;
}

// -------------------------------------------------------------------------
// Fused per-tile kernel: 1 wave handles 16 batch elements end-to-end.
// LDS halves: xact[0,8192) xconv[8192,16384) yact[16384,18432)
// yconv[18432,20480); h2 aliases [0,12800) (x buffers free by then);
// gains (f32) at half-offset 20480. Whole window zero-initialized once so
// unguarded K-slice over-reads are finite.
// -------------------------------------------------------------------------
__global__ __launch_bounds__(32) void fused_kernel(FuseParams pr)
{
  extern __shared__ _Float16 smem[];
  _Float16* xact  = smem;
  _Float16* xconv = smem + 8192;
  _Float16* yact  = smem + 16384;
  _Float16* yconv = smem + 18432;
  _Float16* h2s   = smem;                    // alias: valid after x-path + FC0
  float*    sg    = (float*)(smem + 20480);  // 100 floats of gains

  const int lane = threadIdx.x;
  const int tile = blockIdx.x;

  {
    uint4* z4 = (uint4*)smem;                // 40960 B -> 2560 uint4
    for (int i = lane; i < 2560; i += 32) z4[i] = make_uint4(0u,0u,0u,0u);
  }
  for (int i = lane; i < 100; i += 32) sg[i] = pr.gains[i];
  __syncthreads();

  // ---- Kalman scan + feature build: lanes 0..15, one batch element each ----
  if (lane < 16) {
    const long b = (long)tile*16 + lane;
    float y[2][11];
    for (int c = 0; c < 2; ++c)
      for (int t = 0; t < 11; ++t) y[c][t] = pr.inputy[b*22 + c*11 + t];
    float Hl[8];
    for (int i = 0; i < 8; ++i) Hl[i] = pr.Hm[i];

    float xpo[11][4], xpr[11][4], m1y[11][2];
    for (int m = 0; m < 4; ++m) { float v0 = pr.target[b*44 + m*11]; xpo[0][m] = v0; xpr[0][m] = v0; }
    for (int i = 0; i < 10; ++i) {
      const int t = i + 1;
      const float* g = sg + i*10;
      const float dt = g[0], vv = g[1];
      float xp[4];
      xp[0] = xpo[t-1][0] - dt*(xpo[t-1][1] + vv*xpo[t-1][2] + vv*vv*xpo[t-1][3]);
      xp[1] = xpo[t-1][1]; xp[2] = xpo[t-1][2]; xp[3] = xpo[t-1][3];
      for (int m = 0; m < 4; ++m) xpr[t][m] = xp[m];
      float my0 = Hl[0]*xp[0] + Hl[1]*xp[1] + Hl[2]*xp[2] + Hl[3]*xp[3];
      float my1 = Hl[4]*xp[0] + Hl[5]*xp[1] + Hl[6]*xp[2] + Hl[7]*xp[3];
      m1y[t][0] = my0; m1y[t][1] = my1;
      float e0 = y[0][t] - my0, e1 = y[1][t] - my1;
      for (int m = 0; m < 4; ++m) xpo[t][m] = xp[m] + g[2+m*2]*e0 + g[3+m*2]*e1;
    }
    // features, LDS layout act[n][t*C + ic]
    for (int t = 0; t < 11; ++t) {
      float ev[4] = {0,0,0,0}, up[4] = {0,0,0,0}, ob[2] = {0,0}, iv[2] = {0,0};
      if (t >= 2) {
        for (int m = 0; m < 4; ++m) ev[m] = xpo[t-1][m] - xpo[t-2][m];
        for (int m = 0; m < 4; ++m) up[m] = xpo[t-1][m] - xpr[t-1][m];
        normv(ev, 4); normv(up, 4);
      }
      if (t >= 1) {
        ob[0] = y[0][t] - y[0][t-1];  ob[1] = y[1][t] - y[1][t-1];
        iv[0] = y[0][t] - m1y[t][0];  iv[1] = y[1][t] - m1y[t][1];
        normv(ob, 2); normv(iv, 2);
      }
      for (int m = 0; m < 4; ++m) {
        xact[lane*88 + t*8 + m    ] = (_Float16)ev[m];
        xact[lane*88 + t*8 + 4 + m] = (_Float16)up[m];
      }
      for (int c = 0; c < 2; ++c) {
        yact[lane*44 + t*4 + c    ] = (_Float16)ob[c];
        yact[lane*44 + t*4 + 2 + c] = (_Float16)iv[c];
      }
    }
  }
  __syncthreads();

  // ---- x conv stack: 8->16->32->64->128->256, len 11->9->7->5->3->1 ----
  conv_wmma<  8, 16, 11, 2, true>(pr.wf[0], pr.bs[0], xact, xconv, lane); __syncthreads();
  maxpool_lds< 16, 10>(xconv, xact, lane);                               __syncthreads();
  conv_wmma< 16, 32,  9, 2, true>(pr.wf[1], pr.bs[1], xact, xconv, lane); __syncthreads();
  maxpool_lds< 32,  8>(xconv, xact, lane);                               __syncthreads();
  conv_wmma< 32, 64,  7, 2, true>(pr.wf[2], pr.bs[2], xact, xconv, lane); __syncthreads();
  maxpool_lds< 64,  6>(xconv, xact, lane);                               __syncthreads();
  conv_wmma< 64,128,  5, 2, true>(pr.wf[3], pr.bs[3], xact, xconv, lane); __syncthreads();
  maxpool_lds<128,  4>(xconv, xact, lane);                               __syncthreads();
  conv_wmma<128,256,  3, 2, true>(pr.wf[4], pr.bs[4], xact, xconv, lane); __syncthreads();
  maxpool_lds<256,  2>(xconv, xact, lane);                               __syncthreads();

  // ---- y conv stack: 4->4->8->16->32->64 ----
  conv_wmma<  4,  4, 11, 2, true>(pr.wf[5], pr.bs[5], yact, yconv, lane); __syncthreads();
  maxpool_lds<  4, 10>(yconv, yact, lane);                               __syncthreads();
  conv_wmma<  4,  8,  9, 2, true>(pr.wf[6], pr.bs[6], yact, yconv, lane); __syncthreads();
  maxpool_lds<  8,  8>(yconv, yact, lane);                               __syncthreads();
  conv_wmma<  8, 16,  7, 2, true>(pr.wf[7], pr.bs[7], yact, yconv, lane); __syncthreads();
  maxpool_lds< 16,  6>(yconv, yact, lane);                               __syncthreads();
  conv_wmma< 16, 32,  5, 2, true>(pr.wf[8], pr.bs[8], yact, yconv, lane); __syncthreads();
  maxpool_lds< 32,  4>(yconv, yact, lane);                               __syncthreads();
  conv_wmma< 32, 64,  3, 2, true>(pr.wf[9], pr.bs[9], yact, yconv, lane); __syncthreads();
  maxpool_lds< 64,  2>(yconv, yact, lane);                               __syncthreads();

  // ---- FC0 (256->16) + FC1 (64->4) build z (row stride 20) in yconv ----
  conv_wmma<256, 16, 1, 1, true, 20>(pr.wf[10], pr.bs[10], xact, yconv,      lane); __syncthreads();
  conv_wmma< 64,  4, 1, 1, true, 20>(pr.wf[11], pr.bs[11], yact, yconv + 16, lane); __syncthreads();

  // ---- FC2a (20->800, relu) -> h2 (aliases freed x buffers) ----
  conv_wmma< 20,800, 1, 1, true>(pr.wf[12], pr.bs[12], yconv, h2s, lane); __syncthreads();

  // ---- FC2b (800->8), fp32 straight to global ----
  {
    const _Float16* wf  = pr.wf[13];
    const float*  bias  = pr.bs[13];
    const int n    = lane & 15;
    const int qoff = (lane & 16) ? 16 : 0;
    v8f acc = {0.f,0.f,0.f,0.f,0.f,0.f,0.f,0.f};
    for (int kt = 0; kt < 25; ++kt) {
      v16h a = *(const v16h*)(wf + (size_t)(kt*32 + lane)*16);
      v16h bfrag = *(const v16h*)(h2s + n*800 + kt*32 + qoff);
      acc = __builtin_amdgcn_wmma_f32_16x16x32_f16(false, a, false, bfrag,
                                                   (short)0, acc, false, false);
    }
    if (lane < 16) {  // rows 0..7 == the 8 valid outputs
#pragma unroll
      for (int r = 0; r < 8; ++r)
        pr.out[((size_t)tile*16 + n)*8 + r] = acc[r] + bias[r];
    }
  }
}

// -------------------------------------------------------------------------
extern "C" void kernel_launch(void* const* d_in, const int* in_sizes, int n_in,
                              void* d_out, int out_size, void* d_ws, size_t ws_size,
                              hipStream_t stream)
{
  // {w_idx, b_idx, Cin, Cout, ksz}
  static const int li[14][5] = {
    { 8, 9,   8,  16, 2}, {10,11,  16,  32, 2}, {12,13,  32,  64, 2},
    {14,15,  64, 128, 2}, {16,17, 128, 256, 2},
    {18,19,   4,   4, 2}, {20,21,   4,   8, 2}, {22,23,   8,  16, 2},
    {24,25,  16,  32, 2}, {26,27,  32,  64, 2},
    {28,29, 256,  16, 1}, {30,31,  64,   4, 1},
    {32,33,  20, 800, 1}, {34,35, 800,   8, 1}
  };

  char*  ws    = (char*)d_ws;
  float* gains = (float*)ws;

  kalman_gains_kernel<<<1, 32, 0, stream>>>(
      (const float*)d_in[2], (const float*)d_in[3], (const float*)d_in[4],
      (const float*)d_in[5], (const float*)d_in[6], (const float*)d_in[7], gains);

  FuseParams pr;
  pr.inputy = (const float*)d_in[0];
  pr.target = (const float*)d_in[1];
  pr.Hm     = (const float*)d_in[5];
  pr.gains  = gains;
  pr.out    = (float*)d_out;

  size_t off = 512;  // gains occupy first 400B
  for (int i = 0; i < 14; ++i) {
    const int Cin = li[i][2], Cout = li[i][3], ksz = li[i][4];
    const int ot  = (Cout + 15) / 16;
    const int kt  = (Cin*ksz + 31) / 32;
    const int total = ot*kt*512;
    _Float16* dst = (_Float16*)(ws + off);
    off += (size_t)total * 2;
    swz_kernel<<<(total + 255)/256, 256, 0, stream>>>(
        (const float*)d_in[li[i][0]], dst, Cin, Cout, ksz, kt, total);
    pr.wf[i] = dst;
    pr.bs[i] = (const float*)d_in[li[i][1]];
  }

  const int Bsz   = in_sizes[0] / 22;   // (B,2,11)
  const int tiles = Bsz / 16;
  const size_t smem_bytes = (size_t)20480*2 + 112*4;  // f16 buffers + gains
  fused_kernel<<<tiles, 32, smem_bytes, stream>>>(pr);
}
